// Aggregator_42494406427359
// MI455X (gfx1250) — matrix operand order repaired and checked by reading, not compiled, not run on hardware
//
#include <hip/hip_runtime.h>
#include <hip/hip_bf16.h>

typedef __attribute__((ext_vector_type(16))) _Float16 v16h;
typedef __attribute__((ext_vector_type(8)))  _Float16 v8h;
typedef __attribute__((ext_vector_type(8)))  float    v8f;

#define DIM      64
#define ROWS_PB  128   // rows (edges or nodes) per block
#define NTHREADS 256   // 8 waves of 32

// ---------------------------------------------------------------------------
// WMMA operand layouts per CDNA5 ISA 7.12.2 (v_wmma_f32_16x16x32_f16):
//   A  : lane l -> row M = l&15 ; halves j=0..7  -> K = 8*(l>>4)+j,
//                                 halves j=8..15 -> K = 16+8*(l>>4)+(j-8)
//   B  : lane l -> col N = l&15 ; same K mapping as A
//   C/D: lane l -> col N = l&15 ; vgpr i -> row M = 8*(l>>4)+i
// A-fragments for a wave's 16-row tile are invariant across N-tiles, so they
// are loaded ONCE per layer (both K=32 chunks) and reused for all 4 N-tiles.
// ---------------------------------------------------------------------------
struct AFrag { v16h k0, k1; };   // K = 0..31 and K = 32..63 chunks

__device__ __forceinline__ AFrag load_A64(const _Float16* A, int row, int hp)
{
    union { v16h v; v8h h[2]; } a0, a1;
    const _Float16* ap = A + row * DIM + hp * 8;
    a0.h[0] = *(const v8h*)(ap);
    a0.h[1] = *(const v8h*)(ap + 16);
    a1.h[0] = *(const v8h*)(ap + 32);
    a1.h[1] = *(const v8h*)(ap + 48);
    AFrag f; f.k0 = a0.v; f.k1 = a1.v;
    return f;
}

__device__ __forceinline__ v16h load_B16(const _Float16* Bt, int col, int kt, int hp)
{
    union { v16h v; v8h h[2]; } b;
    const _Float16* bp = Bt + col * DIM + kt * 32 + hp * 8;
    b.h[0] = *(const v8h*)(bp);
    b.h[1] = *(const v8h*)(bp + 16);
    return b.v;
}

__device__ __forceinline__ v8f gemm_tile16(const AFrag& a, const _Float16* Bt,
                                           int col0, float biasn, int rr, int hp)
{
    v8f acc;
#pragma unroll
    for (int i = 0; i < 8; ++i) acc[i] = biasn;
    v16h b0 = load_B16(Bt, col0 + rr, 0, hp);
    v16h b1 = load_B16(Bt, col0 + rr, 1, hp);
    acc = __builtin_amdgcn_wmma_f32_16x16x32_f16(false, a.k0, false, b0,
                                                 (short)0, acc, false, false);
    acc = __builtin_amdgcn_wmma_f32_16x16x32_f16(false, a.k1, false, b1,
                                                 (short)0, acc, false, false);
    return acc;
}

// Load a 64x64 f32 weight matrix from global, convert to f16, store transposed
// into LDS (Wt[n*64+k] = W[k*64+n]). 256 threads x 16 elements.
__device__ __forceinline__ void load_weight_t(const float* __restrict__ W,
                                              _Float16* Wt, int t)
{
#pragma unroll
    for (int j = 0; j < 16; ++j) {
        int idx = t * 16 + j;
        int k = idx >> 6;
        int n = idx & 63;
        Wt[n * DIM + k] = (_Float16)W[idx];
    }
}

// Convert one half-row (32 f32 -> f16) into LDS.
__device__ __forceinline__ void row_f32_to_lds_f16(const float* __restrict__ srcRow,
                                                   _Float16* ldsRow, int cb)
{
    const float4* s4 = (const float4*)srcRow;
#pragma unroll
    for (int j = 0; j < 4; ++j) {
        float4 f0 = s4[(cb >> 2) + 2 * j];
        float4 f1 = s4[(cb >> 2) + 2 * j + 1];
        v8h h = { (_Float16)f0.x, (_Float16)f0.y, (_Float16)f0.z, (_Float16)f0.w,
                  (_Float16)f1.x, (_Float16)f1.y, (_Float16)f1.z, (_Float16)f1.w };
        *(v8h*)(ldsRow + cb + 8 * j) = h;
    }
}

__device__ __forceinline__ void zero_half_row_lds(_Float16* ldsRow, int cb)
{
    v8h zero = {};
#pragma unroll
    for (int j = 0; j < 4; ++j) *(v8h*)(ldsRow + cb + 8 * j) = zero;
}

// ---------------------------------------------------------------------------
// Edge kernel: msg = relu(relu(y[src] @ W1 + b1) @ W2 + b2); z[dst] += msg
// ---------------------------------------------------------------------------
__global__ void __launch_bounds__(NTHREADS)
edge_msg_kernel(const float* __restrict__ y,
                const int*   __restrict__ src,
                const int*   __restrict__ dst,
                const float* __restrict__ W1, const float* __restrict__ b1,
                const float* __restrict__ W2, const float* __restrict__ b2,
                float* __restrict__ z, int E)
{
    __shared__ __align__(16) _Float16 x_lds[ROWS_PB * DIM];
    __shared__ __align__(16) _Float16 h_lds[ROWS_PB * DIM];
    __shared__ __align__(16) _Float16 w1t[DIM * DIM];
    __shared__ __align__(16) _Float16 w2t[DIM * DIM];
    __shared__ float b1_lds[DIM];
    __shared__ float b2_lds[DIM];
    __shared__ int   dst_lds[ROWS_PB];

    const int t     = threadIdx.x;
    const int eBase = blockIdx.x * ROWS_PB;

    load_weight_t(W1, w1t, t);
    load_weight_t(W2, w2t, t);
    if (t < DIM) { b1_lds[t] = b1[t]; b2_lds[t] = b2[t]; }

    // Gather: thread t handles half of row r = t/2 (32 floats).
    {
        const int r  = t >> 1;
        const int cb = (t & 1) * 32;
        const int e  = eBase + r;
        if ((t & 1) == 0) dst_lds[r] = (e < E) ? dst[e] : 0;
        if (e < E) {
            const int node = src[e];
            row_f32_to_lds_f16(y + (size_t)node * DIM, x_lds + r * DIM, cb);
        } else {
            zero_half_row_lds(x_lds + r * DIM, cb);
        }
    }
    __syncthreads();

    const int lane = t & 31;
    const int wv   = t >> 5;       // wave id == M tile
    const int row0 = wv * 16;
    const int rr   = lane & 15;
    const int hp   = lane >> 4;

    // Layer 1: h = relu(x @ W1 + b1)   (each wave writes only its own rows)
    {
        AFrag a = load_A64(x_lds, row0 + rr, hp);
#pragma unroll
        for (int nt = 0; nt < 4; ++nt) {
            const int col0 = nt * 16;
            v8f acc = gemm_tile16(a, w1t, col0, b1_lds[col0 + rr], rr, hp);
#pragma unroll
            for (int i = 0; i < 8; ++i) {
                float v = fmaxf(acc[i], 0.0f);
                h_lds[(row0 + hp * 8 + i) * DIM + col0 + rr] = (_Float16)v;
            }
        }
    }

    // Layer 2 + scatter: z[dst[e]] += relu(h @ W2 + b2)
    {
        AFrag a = load_A64(h_lds, row0 + rr, hp);
#pragma unroll
        for (int nt = 0; nt < 4; ++nt) {
            const int col0 = nt * 16;
            v8f acc = gemm_tile16(a, w2t, col0, b2_lds[col0 + rr], rr, hp);
#pragma unroll
            for (int i = 0; i < 8; ++i) {
                const int m  = row0 + hp * 8 + i;
                const int ee = eBase + m;
                if (ee < E) {
                    float v = fmaxf(acc[i], 0.0f);
                    unsafeAtomicAdd(z + (size_t)dst_lds[m] * DIM + col0 + rr, v);
                }
            }
        }
    }
}

// ---------------------------------------------------------------------------
// Node kernel: h = relu(relu(z @ U1 + c1) @ U2 + c2)
// ---------------------------------------------------------------------------
__global__ void __launch_bounds__(NTHREADS)
node_update_kernel(const float* __restrict__ z,
                   const float* __restrict__ U1, const float* __restrict__ c1,
                   const float* __restrict__ U2, const float* __restrict__ c2,
                   float* __restrict__ out, int N)
{
    __shared__ __align__(16) _Float16 z_lds[ROWS_PB * DIM];
    __shared__ __align__(16) _Float16 h_lds[ROWS_PB * DIM];
    __shared__ __align__(16) _Float16 u1t[DIM * DIM];
    __shared__ __align__(16) _Float16 u2t[DIM * DIM];
    __shared__ float c1_lds[DIM];
    __shared__ float c2_lds[DIM];

    const int t     = threadIdx.x;
    const int nBase = blockIdx.x * ROWS_PB;

    load_weight_t(U1, u1t, t);
    load_weight_t(U2, u2t, t);
    if (t < DIM) { c1_lds[t] = c1[t]; c2_lds[t] = c2[t]; }

    {
        const int r  = t >> 1;
        const int cb = (t & 1) * 32;
        const int n  = nBase + r;
        if (n < N) row_f32_to_lds_f16(z + (size_t)n * DIM, z_lds + r * DIM, cb);
        else       zero_half_row_lds(z_lds + r * DIM, cb);
    }
    __syncthreads();

    const int lane = t & 31;
    const int wv   = t >> 5;
    const int row0 = wv * 16;
    const int rr   = lane & 15;
    const int hp   = lane >> 4;

    // Layer 1: h = relu(z @ U1 + c1)
    {
        AFrag a = load_A64(z_lds, row0 + rr, hp);
#pragma unroll
        for (int nt = 0; nt < 4; ++nt) {
            const int col0 = nt * 16;
            v8f acc = gemm_tile16(a, u1t, col0, c1_lds[col0 + rr], rr, hp);
#pragma unroll
            for (int i = 0; i < 8; ++i) {
                float v = fmaxf(acc[i], 0.0f);
                h_lds[(row0 + hp * 8 + i) * DIM + col0 + rr] = (_Float16)v;
            }
        }
    }

    // Layer 2: out = relu(h @ U2 + c2)
    {
        AFrag a = load_A64(h_lds, row0 + rr, hp);
#pragma unroll
        for (int nt = 0; nt < 4; ++nt) {
            const int col0 = nt * 16;
            v8f acc = gemm_tile16(a, u2t, col0, c2_lds[col0 + rr], rr, hp);
#pragma unroll
            for (int i = 0; i < 8; ++i) {
                const int m = nBase + row0 + hp * 8 + i;
                if (m < N) out[(size_t)m * DIM + col0 + rr] = fmaxf(acc[i], 0.0f);
            }
        }
    }
}

// ---------------------------------------------------------------------------
// Zero workspace accumulator (graph-replay safe: runs every call).
// ---------------------------------------------------------------------------
__global__ void zero_f32_kernel(float4* __restrict__ p, size_t n4)
{
    size_t i      = (size_t)blockIdx.x * blockDim.x + threadIdx.x;
    size_t stride = (size_t)gridDim.x * blockDim.x;
    float4 zero   = make_float4(0.f, 0.f, 0.f, 0.f);
    for (; i < n4; i += stride) p[i] = zero;
}

extern "C" void kernel_launch(void* const* d_in, const int* in_sizes, int n_in,
                              void* d_out, int out_size, void* d_ws, size_t ws_size,
                              hipStream_t stream)
{
    const float* y   = (const float*)d_in[0];
    const int*   src = (const int*)  d_in[1];
    const int*   dst = (const int*)  d_in[2];
    const float* W1  = (const float*)d_in[3];
    const float* b1  = (const float*)d_in[4];
    const float* W2  = (const float*)d_in[5];
    const float* b2  = (const float*)d_in[6];
    const float* U1  = (const float*)d_in[7];
    const float* c1  = (const float*)d_in[8];
    const float* U2  = (const float*)d_in[9];
    const float* c2  = (const float*)d_in[10];
    float* out = (float*)d_out;

    const int N = in_sizes[0] / DIM;   // 50000
    const int E = in_sizes[1];         // 1250000

    float* z = (float*)d_ws;           // [N, 64] f32 accumulator

    // 1) z = 0
    {
        size_t n4 = ((size_t)N * DIM) / 4;
        int blocks = (int)((n4 + 255) / 256);
        if (blocks > 4096) blocks = 4096;
        zero_f32_kernel<<<blocks, 256, 0, stream>>>((float4*)z, n4);
    }
    // 2) edge messages + scatter-add
    {
        int blocks = (E + ROWS_PB - 1) / ROWS_PB;
        edge_msg_kernel<<<blocks, NTHREADS, 0, stream>>>(y, src, dst, W1, b1, W2, b2, z, E);
    }
    // 3) node update
    {
        int blocks = (N + ROWS_PB - 1) / ROWS_PB;
        node_update_kernel<<<blocks, NTHREADS, 0, stream>>>(z, U1, c1, U2, c2, out, N);
    }
}
